// SimpleSSM_62706522521813
// MI455X (gfx1250) — compile-verified, matrix operand order checked by reading, not run on hardware
//
#include <hip/hip_runtime.h>
#include <hip/hip_bf16.h>

// ---------------------------------------------------------------------------
// SimpleSSM on MI455X (gfx1250, wave32, WMMA + Tensor Data Mover)
//   GEMM1 : Bu[m,s]  = sum_i u[b,i,l] * B[s,i]           (m = l*16 + b)
//   SCAN  : x[l]     = A_diag*x[l-1] + Bu[l]             (per (b,s) chain)
//   GEMM2 : Y[b,o,l] = sum_s xs[m,s]*C[o,s] + sum_i uT[m,i]*D[o,i]
//            (fused as one K=1024 GEMM over [xs|uT] x [C|D])
// bf16 A/B operands, f32 accumulation via v_wmma_f32_16x16x32_bf16.
// Weight/activation panels staged into LDS by TENSOR_LOAD_TO_LDS (TDM),
// double-buffered, synchronized with s_wait_tensorcnt.
// ---------------------------------------------------------------------------

#define BZ       16
#define IN_DIM   512
#define S_DIM    512
#define OUT_DIM  512
#define SEQ_L    2048
#define M_TOT    (SEQ_L * BZ)      // 32768

#define BLK_M    128
#define BLK_N    128
#define BLK_K    32
#define LDS_K    40                // 32 + 8 pad (conflict-free b128 frags)

typedef __attribute__((ext_vector_type(16))) __bf16 v16bf;
typedef __attribute__((ext_vector_type(8)))  __bf16 v8bf;
typedef __attribute__((ext_vector_type(8)))  float  v8f;

#if defined(__gfx1250__) && __has_builtin(__builtin_amdgcn_tensor_load_to_lds)
#define HAVE_TDM 1
#else
#define HAVE_TDM 0
#endif

#if HAVE_TDM
typedef __attribute__((ext_vector_type(4))) unsigned int v4u;
typedef __attribute__((ext_vector_type(8))) int          v8i;
typedef __attribute__((ext_vector_type(4))) int          v4i;

// Issue one 2D TDM tile load: rows x cols (bf16 elements), row stride in
// elements; LDS destination uses 40-element padded rows via the D# pad fields
// (pad_interval = 16 DWORDs of data, pad_amount = 4 DWORDs skipped).
__device__ __forceinline__ void tdm_load_2d(unsigned lds_off, const void* gptr,
                                            unsigned rows, unsigned cols,
                                            unsigned row_stride_elems) {
    unsigned long long ga = (unsigned long long)(uintptr_t)gptr;
    v4u g0;
    g0[0] = 1u;                                        // count=1, load, user
    g0[1] = lds_off;                                   // lds_addr (bytes)
    g0[2] = (unsigned)(ga & 0xffffffffu);              // global_addr[31:0]
    g0[3] = (unsigned)((ga >> 32) & 0x1ffffffu) | (2u << 30);  // [56:32]|type=2
    const unsigned td0 = 0x01000000u, td1 = 0x01000000u;   // huge: no OOB clamp
    v8i g1;
    g1[0] = (int)((1u << 16)        // data_size = 2 bytes
                | (1u << 20)        // pad_enable
                | (3u << 22)        // pad_interval: 16 DWORDs (=64B row)
                | (3u << 25));      // pad_amount:   4 DWORDs (=16B skip)
    g1[1] = (int)((td0 & 0xffffu) << 16);                       // tensor_dim0 lo
    g1[2] = (int)(((td0 >> 16) & 0xffffu) | ((td1 & 0xffffu) << 16));
    g1[3] = (int)(((td1 >> 16) & 0xffffu) | (cols << 16));      // tile_dim0
    g1[4] = (int)(rows & 0xffffu);                              // tile_dim1
    g1[5] = (int)row_stride_elems;                              // dim0_stride lo
    g1[6] = 0;
    g1[7] = 0;
    v4i gz4 = {0, 0, 0, 0};
    v8i gz8 = {0, 0, 0, 0, 0, 0, 0, 0};
    __builtin_amdgcn_tensor_load_to_lds(g0, g1, gz4, gz4, gz8, 0);
}
#endif

// Build one 16x32 bf16 WMMA fragment from an LDS row (K-contiguous layout).
// Per ISA 7.12.2: lane half h: elems 0..7 = K h*8.., elems 8..15 = K 16+h*8..
__device__ __forceinline__ v16bf load_frag(const __bf16* row, int h) {
    v8bf lo = *(const v8bf*)(row + h * 8);
    v8bf hi = *(const v8bf*)(row + 16 + h * 8);
    return __builtin_shufflevector(lo, hi, 0,1,2,3,4,5,6,7,8,9,10,11,12,13,14,15);
}

__device__ __forceinline__ v8f wmma_bf16(v16bf a, v16bf b, v8f c) {
    return __builtin_amdgcn_wmma_f32_16x16x32_bf16(
        false, a, false, b, (short)0, c, false, false);
}

// --------------------------- prep kernels ----------------------------------

__global__ void ssm_adiag_kernel(const float* __restrict__ a_unc,
                                 float* __restrict__ ad) {
    int i = blockIdx.x * blockDim.x + threadIdx.x;
    if (i < S_DIM) {
        float x = a_unc[i];
        ad[i] = -(fmaxf(x, 0.0f) + log1pf(expf(-fabsf(x))));
    }
}

__global__ void ssm_cvt_kernel(const float* __restrict__ src,
                               __bf16* __restrict__ dst,
                               int n, int src_cols, int dst_stride, int dst_off) {
    int i = blockIdx.x * blockDim.x + threadIdx.x;
    if (i < n) {
        int r = i / src_cols, c = i - r * src_cols;
        dst[(size_t)r * dst_stride + dst_off + c] = (__bf16)src[i];
    }
}

// --------------------------- GEMM1: Bu = u^T x B^T -------------------------
// A[m,k] = u[b,k,l] gathered+converted by VALU (u is L2-resident: 64MiB<192MB),
// emitting bf16 uT[m,k] as a byproduct. B panel staged by TDM.

__global__ __launch_bounds__(256)
void ssm_gemm1_kernel(const float* __restrict__ u,
                      const __bf16* __restrict__ Bb,
                      float* __restrict__ Bu,
                      __bf16* __restrict__ uT) {
    __shared__ __align__(16) __bf16 ldsA[2][BLK_M][LDS_K];
    __shared__ __align__(16) __bf16 ldsB[2][BLK_N][LDS_K];

    const int tid  = threadIdx.x;
    const int lane = tid & 31;
    const int wave = tid >> 5;
    const int wm   = wave >> 1;          // 0..3 -> rows wm*32
    const int wn   = wave & 1;           // 0..1 -> cols wn*64
    const int lr   = lane & 15;
    const int h    = lane >> 4;

    const int mbase = blockIdx.x * BLK_M;
    const int nbase = blockIdx.y * BLK_N;

    // A loader: one (row, 16-wide K chunk) per thread
    const int arow = tid >> 1;
    const int akq  = (tid & 1) * 16;
    const int amg  = mbase + arow;
    const float* aptr = u + ((size_t)(amg & (BZ - 1)) * IN_DIM) * SEQ_L + (amg >> 4);

    v8f acc[2][4];
    #pragma unroll
    for (int mi = 0; mi < 2; ++mi)
        #pragma unroll
        for (int ni = 0; ni < 4; ++ni) acc[mi][ni] = (v8f){};

    auto stage = [&](int kt, int buf) {
        const int k0 = kt * BLK_K;
        // ---- A: strided f32 gather -> bf16 (and emit uT once) ----
        v8bf v0, v1;
        #pragma unroll
        for (int j = 0; j < 8; ++j) {
            v0[j] = (__bf16)aptr[(size_t)(k0 + akq + j) * SEQ_L];
            v1[j] = (__bf16)aptr[(size_t)(k0 + akq + 8 + j) * SEQ_L];
        }
        *(v8bf*)&ldsA[buf][arow][akq]     = v0;
        *(v8bf*)&ldsA[buf][arow][akq + 8] = v1;
        if (blockIdx.y == 0) {
            *(v8bf*)(uT + (size_t)amg * IN_DIM + k0 + akq)     = v0;
            *(v8bf*)(uT + (size_t)amg * IN_DIM + k0 + akq + 8) = v1;
        }
        // ---- B panel: TDM async tile load (wave 0 issues) ----
#if HAVE_TDM
        if (wave == 0)
            tdm_load_2d((unsigned)(uintptr_t)&ldsB[buf][0][0],
                        Bb + (size_t)nbase * IN_DIM + k0,
                        BLK_N, BLK_K, IN_DIM);
#else
        const __bf16* bp = Bb + (size_t)(nbase + (tid >> 1)) * IN_DIM + k0 + akq;
        *(v8bf*)&ldsB[buf][tid >> 1][akq]     = *(const v8bf*)bp;
        *(v8bf*)&ldsB[buf][tid >> 1][akq + 8] = *(const v8bf*)(bp + 8);
#endif
    };

    stage(0, 0);
#if HAVE_TDM
    if (wave == 0) __builtin_amdgcn_s_wait_tensorcnt(0);
#endif
    __syncthreads();

    const int KT = IN_DIM / BLK_K;       // 16
    for (int kt = 0; kt < KT; ++kt) {
        const int buf = kt & 1;
        if (kt + 1 < KT) stage(kt + 1, buf ^ 1);

        v16bf a0 = load_frag(&ldsA[buf][wm * 32 + lr][0],      h);
        v16bf a1 = load_frag(&ldsA[buf][wm * 32 + 16 + lr][0], h);
        #pragma unroll
        for (int ni = 0; ni < 4; ++ni) {
            v16bf bf = load_frag(&ldsB[buf][wn * 64 + ni * 16 + lr][0], h);
            acc[0][ni] = wmma_bf16(a0, bf, acc[0][ni]);
            acc[1][ni] = wmma_bf16(a1, bf, acc[1][ni]);
        }
#if HAVE_TDM
        if (wave == 0) __builtin_amdgcn_s_wait_tensorcnt(0);
#endif
        __syncthreads();
    }

    // ---- epilogue: Bu[m][s] f32, coalesced across lanes ----
    #pragma unroll
    for (int mi = 0; mi < 2; ++mi)
        #pragma unroll
        for (int ni = 0; ni < 4; ++ni) {
            int n = nbase + wn * 64 + ni * 16 + lr;
            int m = mbase + wm * 32 + mi * 16 + h * 8;
            #pragma unroll
            for (int r = 0; r < 8; ++r)
                Bu[(size_t)(m + r) * S_DIM + n] = acc[mi][ni][r];
        }
}

// --------------------------- SCAN ------------------------------------------
__global__ __launch_bounds__(256)
void ssm_scan_kernel(const float* __restrict__ Bu,
                     const float* __restrict__ ad,
                     const float* __restrict__ h0,
                     __bf16* __restrict__ xs) {
    int t = blockIdx.x * blockDim.x + threadIdx.x;   // 0..8191
    int s = t & (S_DIM - 1);
    int b = t >> 9;
    float a = ad[s];
    float x = h0[s];
    const float* bu = Bu + (size_t)b * S_DIM + s;
    __bf16* xo = xs + (size_t)b * S_DIM + s;
    const size_t stride = (size_t)BZ * S_DIM;        // 8192
    for (int l = 0; l < SEQ_L; ++l) {
        if (l + 8 < SEQ_L)
            __builtin_prefetch(bu + (size_t)(l + 8) * stride, 0, 0);
        x = fmaf(x, a, bu[(size_t)l * stride]);
        xo[(size_t)l * stride] = (__bf16)x;
    }
}

// --------------------------- GEMM2: Y = [xs|uT] x [C|D]^T ------------------
// Both A (activations, bf16) and B (weights, bf16) panels staged by TDM.

__global__ __launch_bounds__(256)
void ssm_gemm2_kernel(const __bf16* __restrict__ xs,
                      const __bf16* __restrict__ uT,
                      const __bf16* __restrict__ CDb,   // [o][k], stride 1024
                      float* __restrict__ out) {
    __shared__ __align__(16) __bf16 ldsA[2][BLK_M][LDS_K];
    __shared__ __align__(16) __bf16 ldsB[2][BLK_N][LDS_K];

    const int tid  = threadIdx.x;
    const int lane = tid & 31;
    const int wave = tid >> 5;
    const int wm   = wave >> 1;
    const int wn   = wave & 1;
    const int lr   = lane & 15;
    const int h    = lane >> 4;

    const int mbase = blockIdx.x * BLK_M;
    const int nbase = blockIdx.y * BLK_N;

    v8f acc[2][4];
    #pragma unroll
    for (int mi = 0; mi < 2; ++mi)
        #pragma unroll
        for (int ni = 0; ni < 4; ++ni) acc[mi][ni] = (v8f){};

    auto stage = [&](int kt, int buf) {
        const int k0 = kt * BLK_K;
        const __bf16* asrc = (k0 < S_DIM)
            ? xs + (size_t)mbase * S_DIM + k0
            : uT + (size_t)mbase * IN_DIM + (k0 - S_DIM);
#if HAVE_TDM
        if (wave == 0) {
            tdm_load_2d((unsigned)(uintptr_t)&ldsA[buf][0][0], asrc,
                        BLK_M, BLK_K, S_DIM);
            tdm_load_2d((unsigned)(uintptr_t)&ldsB[buf][0][0],
                        CDb + (size_t)nbase * 1024 + k0,
                        BLK_N, BLK_K, 1024);
        }
#else
        const int row = tid >> 1, kq = (tid & 1) * 16;
        const __bf16* ap = asrc + (size_t)row * S_DIM + kq;
        *(v8bf*)&ldsA[buf][row][kq]     = *(const v8bf*)ap;
        *(v8bf*)&ldsA[buf][row][kq + 8] = *(const v8bf*)(ap + 8);
        const __bf16* bp = CDb + (size_t)(nbase + row) * 1024 + k0 + kq;
        *(v8bf*)&ldsB[buf][row][kq]     = *(const v8bf*)bp;
        *(v8bf*)&ldsB[buf][row][kq + 8] = *(const v8bf*)(bp + 8);
#endif
    };

    stage(0, 0);
#if HAVE_TDM
    if (wave == 0) __builtin_amdgcn_s_wait_tensorcnt(0);
#endif
    __syncthreads();

    const int KT = 1024 / BLK_K;          // 32
    for (int kt = 0; kt < KT; ++kt) {
        const int buf = kt & 1;
        if (kt + 1 < KT) stage(kt + 1, buf ^ 1);

        v16bf a0 = load_frag(&ldsA[buf][wm * 32 + lr][0],      h);
        v16bf a1 = load_frag(&ldsA[buf][wm * 32 + 16 + lr][0], h);
        #pragma unroll
        for (int ni = 0; ni < 4; ++ni) {
            v16bf bf = load_frag(&ldsB[buf][wn * 64 + ni * 16 + lr][0], h);
            acc[0][ni] = wmma_bf16(a0, bf, acc[0][ni]);
            acc[1][ni] = wmma_bf16(a1, bf, acc[1][ni]);
        }
#if HAVE_TDM
        if (wave == 0) __builtin_amdgcn_s_wait_tensorcnt(0);
#endif
        __syncthreads();
    }

    // epilogue: Y[b, o, l]; m = l*16 + b
    #pragma unroll
    for (int mi = 0; mi < 2; ++mi)
        #pragma unroll
        for (int ni = 0; ni < 4; ++ni) {
            int o = nbase + wn * 64 + ni * 16 + lr;
            int m = mbase + wm * 32 + mi * 16 + h * 8;
            #pragma unroll
            for (int r = 0; r < 8; ++r) {
                int mg = m + r;
                out[(size_t)(mg & (BZ - 1)) * OUT_DIM * SEQ_L
                    + (size_t)o * SEQ_L + (mg >> 4)] = acc[mi][ni][r];
            }
        }
}

// --------------------------- launch ----------------------------------------

extern "C" void kernel_launch(void* const* d_in, const int* in_sizes, int n_in,
                              void* d_out, int out_size, void* d_ws, size_t ws_size,
                              hipStream_t stream) {
    const float* u     = (const float*)d_in[0];
    const float* a_unc = (const float*)d_in[1];
    const float* Bmat  = (const float*)d_in[2];
    const float* Cmat  = (const float*)d_in[3];
    const float* Dmat  = (const float*)d_in[4];
    const float* h0    = (const float*)d_in[5];
    float* out = (float*)d_out;

    char* w = (char*)d_ws;
    float*  ad  = (float*)w;                               // 2 KiB
    __bf16* Bb  = (__bf16*)(w + 4096);                     // 512 KiB
    __bf16* CDb = (__bf16*)(w + 4096 + (512 << 10));       // 1 MiB
    char* w2 = w + 4096 + (512 << 10) + (1 << 20);
    float*  Bu  = (float*)w2;                              // 64 MiB
    __bf16* xsb = (__bf16*)(w2 + ((size_t)64 << 20));      // 32 MiB
    __bf16* uT  = (__bf16*)(w2 + ((size_t)96 << 20));      // 32 MiB

    ssm_adiag_kernel<<<2, 256, 0, stream>>>(a_unc, ad);
    const int nW = S_DIM * IN_DIM;
    ssm_cvt_kernel<<<(nW + 255) / 256, 256, 0, stream>>>(Bmat, Bb,  nW, IN_DIM, 512,  0);
    ssm_cvt_kernel<<<(nW + 255) / 256, 256, 0, stream>>>(Cmat, CDb, nW, S_DIM, 1024, 0);
    ssm_cvt_kernel<<<(nW + 255) / 256, 256, 0, stream>>>(Dmat, CDb, nW, IN_DIM, 1024, 512);

    ssm_gemm1_kernel<<<dim3(M_TOT / BLK_M, S_DIM / BLK_N), 256, 0, stream>>>(u, Bb, Bu, uT);
    ssm_scan_kernel<<<(BZ * S_DIM) / 256, 256, 0, stream>>>(Bu, ad, h0, xsb);
    ssm_gemm2_kernel<<<dim3(M_TOT / BLK_M, OUT_DIM / BLK_N), 256, 0, stream>>>(xsb, uT, CDb, out);
}
